// MultiScaleDecoder_slow_nonAuto_11373073400243
// MI455X (gfx1250) — matrix operand order repaired and checked by reading, not compiled, not run on hardware
//
#include <hip/hip_runtime.h>
#include <math.h>

typedef __attribute__((ext_vector_type(16))) __bf16 v16bf;
typedef __attribute__((ext_vector_type(8)))  float  v8f;
typedef unsigned short u16;

union Frag { uint4 u[2]; v16bf v; };

__device__ __forceinline__ u16 f2bf(float f) {
  unsigned u = __float_as_uint(f);
  u += 0x7fffu + ((u >> 16) & 1u);       // round-to-nearest-even
  return (u16)(u >> 16);
}
__device__ __forceinline__ float sigmoidf_(float x) { return 1.0f / (1.0f + __expf(-x)); }
// overflow-safe fast tanh: tanh(|x|) = 1 - 2t/(1+t), t = exp(-2|x|) in (0,1]
__device__ __forceinline__ float tanhf_(float x) {
  float t = __expf(-2.0f * fabsf(x));
  float y = 1.0f - 2.0f * t / (1.0f + t);
  return copysignf(y, x);
}

// ---------------------------------------------------------------- cvt f32->bf16
__global__ void cvt_f32_bf16(const float* __restrict__ src, u16* __restrict__ dst, int n) {
  int i = blockIdx.x * blockDim.x + threadIdx.x;
  if (i < n) dst[i] = f2bf(src[i]);
}

// ---------------------------------------------------------------- build slow_input = [context | e] in bf16
__global__ void build_slow(const float* __restrict__ ctx, const float* __restrict__ e,
                           u16* __restrict__ slow) {
  int i = blockIdx.x * blockDim.x + threadIdx.x;   // 256*640 total
  if (i < 256 * 640) {
    int b = i / 640, j = i % 640;
    float v = (j < 512) ? ctx[b * 512 + j] : e[b * 128 + (j - 512)];
    slow[i] = f2bf(v);
  }
}

// ---------------------------------------------------------------- generic WMMA GEMM: out = act(X(MxK) @ W(NxK)^T + bias)
// one wave per 16x16 output tile, 8 waves per block; M,N multiples of 16, K multiple of 32
__global__ __launch_bounds__(256) void gemm_bf16_nt(
    const u16* __restrict__ X, const u16* __restrict__ W,
    const float* __restrict__ bias, float* __restrict__ out,
    int M, int N, int K, int do_relu)
{
  const int lane = threadIdx.x & 31;
  const int wave = threadIdx.x >> 5;
  const int ln = lane & 15;
  const int kh = lane >> 4;
  const int ntn = N >> 4;
  const int tile = blockIdx.x * (blockDim.x >> 5) + wave;
  const int m0 = (tile / ntn) << 4;
  const int n0 = (tile % ntn) << 4;

  const u16* xrow = X + (size_t)(m0 + ln) * K;
  const u16* wrow = W + (size_t)(n0 + ln) * K;

  v8f acc = {};
  for (int kb = 0; kb < K; kb += 32) {
    Frag a, b;
    // A 16x32 bf16: lane = row, K-half selected by kh
    a.u[0] = *(const uint4*)(xrow + kb + 8 * kh);
    a.u[1] = *(const uint4*)(xrow + kb + 16 + 8 * kh);
    // B 32x16 bf16: lane = column, K-half selected by kh (same order as A)
    b.u[0] = *(const uint4*)(wrow + kb + 8 * kh);
    b.u[1] = *(const uint4*)(wrow + kb + 16 + 8 * kh);
    acc = __builtin_amdgcn_wmma_f32_16x16x32_bf16(false, a.v, false, b.v,
                                                  (short)0, acc, false, false);
  }
  const int n = n0 + ln;
  const float bv = bias ? bias[n] : 0.0f;
#pragma unroll
  for (int g = 0; g < 8; ++g) {
    const int m = m0 + g + 8 * kh;     // C/D layout: M = g + 8*(lane>>4), N = lane&15
    float v = acc[g] + bv;
    if (do_relu) v = fmaxf(v, 0.0f);
    out[(size_t)m * N + n] = v;
  }
}

// ---------------------------------------------------------------- persistent GRU recurrence
// 16 blocks x 512 threads (16 waves); block handles 16 batch rows for all 512 steps.
// Wave w owns H-columns [32w, 32w+32) for the gate math; waves 0-7 own D-columns [16w,16w+16) for FC.
#define HSTR 520   // LDS row stride (u16) for h: row->row bank rotation avoids conflicts

__global__ __launch_bounds__(512) void gru_seq(
    const u16*  __restrict__ Whh,    // 1536 x 512 bf16
    const u16*  __restrict__ Wfc,    // 128 x 512 bf16
    const float* __restrict__ gi,    // 256 x 1536  (= slow_input @ W_ih^T + b_ih)
    const float* __restrict__ b_hh,  // 1536
    const float* __restrict__ b_fc,  // 128
    const float* __restrict__ z_ctx, // 256 x 128 x 64
    float* __restrict__ out)         // 256 x 128 x 512
{
  __shared__ u16 h_bf[16 * HSTR];

  const int tid  = threadIdx.x;
  const int lane = tid & 31;
  const int wave = tid >> 5;          // 0..15
  const int ln   = lane & 15;
  const int kh   = lane >> 4;
  const int b0   = blockIdx.x * 16;
  const int c0   = wave * 32;         // this wave's H-column block

  for (int i = tid; i < 16 * HSTR; i += blockDim.x) h_bf[i] = 0;   // h0 = 0

  // Preload step-invariant inputs into registers (accumulator layout) + W_hh row pointers.
  float gi_reg[2][3][8];
  float bhh_reg[2][3];
  const u16* wp[2][3];
#pragma unroll
  for (int tt = 0; tt < 2; ++tt)
#pragma unroll
    for (int gate = 0; gate < 3; ++gate) {
      const int col = gate * 512 + c0 + 16 * tt + ln;
      bhh_reg[tt][gate] = b_hh[col];
      wp[tt][gate] = Whh + (size_t)col * 512;
#pragma unroll
      for (int g = 0; g < 8; ++g)
        gi_reg[tt][gate][g] = gi[(size_t)(b0 + g + 8 * kh) * 1536 + col];
    }

  float h_reg[2][8];                  // carried f32 hidden state (this wave's 16x32 block)
#pragma unroll
  for (int tt = 0; tt < 2; ++tt)
#pragma unroll
    for (int g = 0; g < 8; ++g) h_reg[tt][g] = 0.0f;

  // FC state (waves 0-7 only): persistent accumulator zh_t = zh_{t-1} + h_t @ Wfc^T + b_fc
  const int d = (wave & 7) * 16 + ln;
  v8f zp = {};
  float bfc = 0.0f;
  if (wave < 8) {
#pragma unroll
    for (int g = 0; g < 8; ++g)
      zp[g] = z_ctx[((size_t)(b0 + g + 8 * kh) * 128 + d) * 64 + 63];
    bfc = b_fc[d];
  }
  const u16* wf = Wfc + (size_t)d * 512;

  __syncthreads();

  for (int t = 0; t < 512; ++t) {
    // ---- gh = h @ Whh^T : 6 independent accumulator chains, A fragment loaded once per kb ----
    v8f acc[2][3];
#pragma unroll
    for (int tt = 0; tt < 2; ++tt)
#pragma unroll
      for (int gate = 0; gate < 3; ++gate) { v8f z = {}; acc[tt][gate] = z; }

    for (int kb = 0; kb < 512; kb += 32) {
      Frag a;
      const u16* hrow = &h_bf[ln * HSTR + kb];
      a.u[0] = *(const uint4*)(hrow + 8 * kh);
      a.u[1] = *(const uint4*)(hrow + 16 + 8 * kh);
#pragma unroll
      for (int tt = 0; tt < 2; ++tt)
#pragma unroll
        for (int gate = 0; gate < 3; ++gate) {
          Frag b;
          b.u[0] = *(const uint4*)(wp[tt][gate] + kb + 8 * kh);
          b.u[1] = *(const uint4*)(wp[tt][gate] + kb + 16 + 8 * kh);
          acc[tt][gate] = __builtin_amdgcn_wmma_f32_16x16x32_bf16(
              false, a.v, false, b.v, (short)0, acc[tt][gate], false, false);
        }
    }
    // gates in f32 (b_hh_n stays inside the r* term, matching the reference)
#pragma unroll
    for (int tt = 0; tt < 2; ++tt)
#pragma unroll
      for (int g = 0; g < 8; ++g) {
        const float r  = sigmoidf_(gi_reg[tt][0][g] + acc[tt][0][g] + bhh_reg[tt][0]);
        const float zg = sigmoidf_(gi_reg[tt][1][g] + acc[tt][1][g] + bhh_reg[tt][1]);
        const float nn = tanhf_(gi_reg[tt][2][g] + r * (acc[tt][2][g] + bhh_reg[tt][2]));
        h_reg[tt][g] = (1.0f - zg) * nn + zg * h_reg[tt][g];
      }

    __syncthreads();                       // everyone done reading h_{t-1}
#pragma unroll
    for (int tt = 0; tt < 2; ++tt) {
      const int col = c0 + 16 * tt + ln;
#pragma unroll
      for (int g = 0; g < 8; ++g)
        h_bf[(g + 8 * kh) * HSTR + col] = f2bf(h_reg[tt][g]);
    }
    __syncthreads();                       // h_t visible to all waves

    // ---- zh accumulation: zp += h_t @ Wfc^T (two independent WMMA chains for ILP) ----
    if (wave < 8) {
      v8f acc2 = {};
      for (int kb = 0; kb < 512; kb += 64) {
        Frag a0, b0, a1, b1;
        const u16* hrow0 = &h_bf[ln * HSTR + kb];
        const u16* hrow1 = hrow0 + 32;
        a0.u[0] = *(const uint4*)(hrow0 + 8 * kh);
        a0.u[1] = *(const uint4*)(hrow0 + 16 + 8 * kh);
        b0.u[0] = *(const uint4*)(wf + kb + 8 * kh);
        b0.u[1] = *(const uint4*)(wf + kb + 16 + 8 * kh);
        zp = __builtin_amdgcn_wmma_f32_16x16x32_bf16(false, a0.v, false, b0.v, (short)0, zp, false, false);
        a1.u[0] = *(const uint4*)(hrow1 + 8 * kh);
        a1.u[1] = *(const uint4*)(hrow1 + 16 + 8 * kh);
        b1.u[0] = *(const uint4*)(wf + kb + 32 + 8 * kh);
        b1.u[1] = *(const uint4*)(wf + kb + 32 + 16 + 8 * kh);
        acc2 = __builtin_amdgcn_wmma_f32_16x16x32_bf16(false, a1.v, false, b1.v, (short)0, acc2, false, false);
      }
#pragma unroll
      for (int g = 0; g < 8; ++g) {
        zp[g] += acc2[g] + bfc;
        out[((size_t)(b0 + g + 8 * kh) * 128 + d) * 512 + t] = zp[g];
      }
    }
  }
}

// ----------------------------------------------------------------
extern "C" void kernel_launch(void* const* d_in, const int* in_sizes, int n_in,
                              void* d_out, int out_size, void* d_ws, size_t ws_size,
                              hipStream_t stream) {
  const float* z_ctx = (const float*)d_in[0];
  // d_in[1..3] = z, z_slow, z_fast: only shapes matter, unused
  const float* e     = (const float*)d_in[4];
  const float* W_enc = (const float*)d_in[5];
  const float* b_enc = (const float*)d_in[6];
  const float* W_ih  = (const float*)d_in[7];
  const float* W_hh  = (const float*)d_in[8];
  const float* b_ih  = (const float*)d_in[9];
  const float* b_hh  = (const float*)d_in[10];
  const float* W_fc  = (const float*)d_in[11];
  const float* b_fc  = (const float*)d_in[12];
  float* out = (float*)d_out;

  char* ws = (char*)d_ws;
  size_t off = 0;
  auto alloc = [&](size_t bytes) -> void* {
    off = (off + 255) & ~(size_t)255;
    void* p = (void*)(ws + off);
    off += bytes;
    return p;
  };

  u16*   zctx_bf = (u16*)  alloc((size_t)256 * 8192 * 2);
  u16*   Wenc_bf = (u16*)  alloc((size_t)512 * 8192 * 2);
  u16*   Wih_bf  = (u16*)  alloc((size_t)1536 * 640 * 2);
  u16*   Whh_bf  = (u16*)  alloc((size_t)1536 * 512 * 2);
  u16*   Wfc_bf  = (u16*)  alloc((size_t)128 * 512 * 2);
  float* ctx     = (float*)alloc((size_t)256 * 512 * 4);
  u16*   slow_bf = (u16*)  alloc((size_t)256 * 640 * 2);
  float* gi      = (float*)alloc((size_t)256 * 1536 * 4);

  int n;
  n = 256 * 8192;  cvt_f32_bf16<<<(n + 255) / 256, 256, 0, stream>>>(z_ctx, zctx_bf, n);
  n = 512 * 8192;  cvt_f32_bf16<<<(n + 255) / 256, 256, 0, stream>>>(W_enc, Wenc_bf, n);
  n = 1536 * 640;  cvt_f32_bf16<<<(n + 255) / 256, 256, 0, stream>>>(W_ih, Wih_bf, n);
  n = 1536 * 512;  cvt_f32_bf16<<<(n + 255) / 256, 256, 0, stream>>>(W_hh, Whh_bf, n);
  n = 128 * 512;   cvt_f32_bf16<<<(n + 255) / 256, 256, 0, stream>>>(W_fc, Wfc_bf, n);

  // context = relu(z_ctx_flat @ W_enc^T + b_enc):  M=256 N=512 K=8192 -> 512 tiles / 8 waves
  gemm_bf16_nt<<<512 / 8, 256, 0, stream>>>(zctx_bf, Wenc_bf, b_enc, ctx, 256, 512, 8192, 1);

  n = 256 * 640;
  build_slow<<<(n + 255) / 256, 256, 0, stream>>>(ctx, e, slow_bf);

  // gi = slow_input @ W_ih^T + b_ih:  M=256 N=1536 K=640 -> 1536 tiles / 8 waves
  gemm_bf16_nt<<<1536 / 8, 256, 0, stream>>>(slow_bf, Wih_bf, b_ih, gi, 256, 1536, 640, 0);

  // persistent GRU: 16 batch-blocks x 16 waves, each block runs all 512 steps
  gru_seq<<<16, 512, 0, stream>>>(Whh_bf, Wfc_bf, gi, b_hh, b_fc, z_ctx, out);
}
